// CLUST_GCN_VS_73555609911566
// MI455X (gfx1250) — matrix operand order repaired and checked by reading, not compiled, not run on hardware
//
#include <hip/hip_runtime.h>
#include <hip/hip_bf16.h>
#include <cstdint>

#define GN 100000
#define GE 1600000
#define HID 64
#define OUTC 32
#define NEG_SLOPE 0.2f

typedef __attribute__((ext_vector_type(16))) _Float16 v16h;
typedef __attribute__((ext_vector_type(8)))  _Float16 v8h;
typedef __attribute__((ext_vector_type(8)))  float    v8f;

// ---------------------------------------------------------------- small utils

__global__ void zero_int(int* __restrict__ p, int n) {
    int i = blockIdx.x * blockDim.x + threadIdx.x;
    if (i < n) p[i] = 0;
}

__global__ void copy_int(const int* __restrict__ a, int* __restrict__ b, int n) {
    int i = blockIdx.x * blockDim.x + threadIdx.x;
    if (i < n) b[i] = a[i];
}

__device__ __forceinline__ float leaky(float x) {
    return x > 0.0f ? x : NEG_SLOPE * x;
}

// ---------------------------------------------------------------- CSR build (built once, reused by all 3 layers)

__global__ void degree_count(const int* __restrict__ dst, int* __restrict__ cnt, int e) {
    int i = blockIdx.x * blockDim.x + threadIdx.x;
    if (i < e) atomicAdd(&cnt[dst[i]], 1);
}

// single-block 2-level exclusive scan: 1024 threads, each owns a serial chunk
__global__ void scan_kernel(const int* __restrict__ cnt, int* __restrict__ rowstart, int n) {
    __shared__ int sums[1024];
    int t = threadIdx.x;
    int chunk = (n + 1023) >> 10;
    int lo = t * chunk;
    int hi = min(lo + chunk, n);
    int own = 0;
    for (int i = lo; i < hi; i++) own += cnt[i];
    sums[t] = own;
    __syncthreads();
#pragma unroll
    for (int off = 1; off < 1024; off <<= 1) {
        int v = (t >= off) ? sums[t - off] : 0;
        __syncthreads();
        sums[t] += v;
        __syncthreads();
    }
    int run = sums[t] - own;  // exclusive prefix at this chunk's start
    for (int i = lo; i < hi; i++) { rowstart[i] = run; run += cnt[i]; }
    if (t == 0) rowstart[n] = sums[1023];
}

__global__ void csr_fill(const int* __restrict__ src, const int* __restrict__ dst,
                         int* __restrict__ cursor, int* __restrict__ csr_src, int e) {
    int i = blockIdx.x * blockDim.x + threadIdx.x;
    if (i >= e) return;
    int pos = atomicAdd(&cursor[dst[i]], 1);
    csr_src[pos] = src[i];
}

__global__ void deginv_kernel(const int* __restrict__ cnt, float* __restrict__ deg_inv, int n) {
    int i = blockIdx.x * blockDim.x + threadIdx.x;
    if (i < n) deg_inv[i] = 1.0f / ((float)cnt[i] + 1.0f);  // +1 for self loop
}

// ---------------------------------------------------------------- GCN gather (atomic-free)
// wave per node; lane f accumulates features f, f+32; 32 edge indices batched into
// lanes then shuffle-broadcast so every x[src] row read is a coalesced 128B burst.

__global__ void __launch_bounds__(256)
gcn_gather(const float* __restrict__ x, const int* __restrict__ csr_src,
           const int* __restrict__ rowstart, const float* __restrict__ deg_inv,
           _Float16* __restrict__ actA, _Float16* __restrict__ actB, int n) {
    int gid  = blockIdx.x * blockDim.x + threadIdx.x;
    int wid  = gid >> 5;
    int lane = threadIdx.x & 31;
    if (wid >= n) return;
    int s0 = rowstart[wid], s1 = rowstart[wid + 1];
    float a0 = 0.0f, a1 = 0.0f;
    for (int j0 = s0; j0 < s1; j0 += 32) {
        int idx = (j0 + lane < s1) ? csr_src[j0 + lane] : 0;
        int m = min(32, s1 - j0);
        for (int k = 0; k < m; k++) {
            int s = __shfl(idx, k, 32);
            const float* xs = x + (size_t)s * HID;
            a0 += xs[lane];
            a1 += xs[lane + 32];
        }
    }
    float di  = deg_inv[wid];
    const float* xi = x + (size_t)wid * HID;
    float xv0 = xi[lane], xv1 = xi[lane + 32];
    _Float16* oa = actA + (size_t)wid * HID;
    _Float16* ob = actB + (size_t)wid * HID;
    oa[lane]      = (_Float16)((a0 + xv0) * di);
    oa[lane + 32] = (_Float16)((a1 + xv1) * di);
    ob[lane]      = (_Float16)xv0;
    ob[lane + 32] = (_Float16)xv1;
}

// ---------------------------------------------------------------- weight packing into WMMA B-fragment layout
// fragment (kt, nt): 32 lanes x 16 halves; lanes 0-15 => K = kt*32 + 0..15,
// lanes 16-31 => K = kt*32 + 16..31; element e -> K = base + e; col = nt*16 + (lane & 15)

__global__ void pack_weights(const float* __restrict__ Wout, const float* __restrict__ Wroot,
                             const float* __restrict__ W1, const float* __restrict__ W2,
                             _Float16* __restrict__ wp) {
    int idx = blockIdx.x * blockDim.x + threadIdx.x;
    const float* W; int NT, Ncols, base;
    if      (idx < 4096)  { W = Wout;  NT = 4; Ncols = 64; base = 0;     }
    else if (idx < 8192)  { W = Wroot; NT = 4; Ncols = 64; base = 4096;  }
    else if (idx < 12288) { W = W1;    NT = 4; Ncols = 64; base = 8192;  }
    else if (idx < 14336) { W = W2;    NT = 2; Ncols = 32; base = 12288; }
    else return;
    int li   = idx - base;
    int e    = li & 15;
    int lane = (li >> 4) & 31;
    int frag = li >> 9;
    int nt   = frag % NT;
    int kt   = frag / NT;
    int k    = kt * 32 + (lane >> 4) * 16 + e;
    int n    = nt * 16 + (lane & 15);
    wp[idx] = (_Float16)W[k * Ncols + n];
}

// ---------------------------------------------------------------- WMMA fragment loaders

// A 16x32 fragment, contiguous k order (matches packed B): lanes 0-15 row M=lane, halves 0..15 = K 0..15
//                        lanes 16-31 row M=lane-16, K 16..31
__device__ __forceinline__ v16h load_a_frag(const _Float16* __restrict__ act, int row0, int kt, int lane) {
    int l  = lane & 15;
    int hb = lane >> 4;
    const _Float16* p = act + (size_t)(row0 + l) * HID + kt * 32 + hb * 16;
    return *(const v16h*)(p);
}

__device__ __forceinline__ v16h load_b_frag(const _Float16* __restrict__ wp, int kt, int nt, int NT, int lane) {
    return *(const v16h*)(wp + (((size_t)(kt * NT + nt) * 32 + lane) << 4));
}

#define WMMA_F16(a, b, c) \
    __builtin_amdgcn_wmma_f32_16x16x32_f16(false, (a), false, (b), (short)0, (c), false, false)

// ---------------------------------------------------------------- GEMM kernels (1 wave = 16-row output tile)

// h1 = relu(aggs @ W_out + x @ W_root + b_out) -> f16 (in place over this tile's actA rows)
__global__ void __launch_bounds__(256)
gemm_gcn(const _Float16* __restrict__ actA, const _Float16* __restrict__ actB,
         const _Float16* __restrict__ wpOut, const _Float16* __restrict__ wpRoot,
         const float* __restrict__ bias, _Float16* __restrict__ outH, int ntiles) {
    int gid  = blockIdx.x * blockDim.x + threadIdx.x;
    int wid  = gid >> 5;
    int lane = threadIdx.x & 31;
    if (wid >= ntiles) return;
    int row0 = wid * 16;

    v8f acc[4];
    v8f z = {};
#pragma unroll
    for (int nt = 0; nt < 4; nt++) acc[nt] = z;

#pragma unroll
    for (int kt = 0; kt < 2; kt++) {
        v16h a = load_a_frag(actA, row0, kt, lane);
#pragma unroll
        for (int nt = 0; nt < 4; nt++)
            acc[nt] = WMMA_F16(a, load_b_frag(wpOut, kt, nt, 4, lane), acc[nt]);
    }
#pragma unroll
    for (int kt = 0; kt < 2; kt++) {
        v16h a = load_a_frag(actB, row0, kt, lane);
#pragma unroll
        for (int nt = 0; nt < 4; nt++)
            acc[nt] = WMMA_F16(a, load_b_frag(wpRoot, kt, nt, 4, lane), acc[nt]);
    }

    int l  = lane & 15;
    int hb = lane >> 4;
#pragma unroll
    for (int nt = 0; nt < 4; nt++) {
        float bv = bias[nt * 16 + l];
#pragma unroll
        for (int r = 0; r < 8; r++) {
            int row = row0 + hb * 8 + r;
            float v = acc[nt][r] + bv;
            v = v > 0.0f ? v : 0.0f;
            outH[(size_t)row * HID + nt * 16 + l] = (_Float16)v;
        }
    }
}

// h = act @ W -> fp32 (no bias; GAT adds bias after aggregation)
template <int NT>
__global__ void __launch_bounds__(256)
gemm_gat(const _Float16* __restrict__ act, const _Float16* __restrict__ wp,
         float* __restrict__ outF, int ntiles) {
    int gid  = blockIdx.x * blockDim.x + threadIdx.x;
    int wid  = gid >> 5;
    int lane = threadIdx.x & 31;
    if (wid >= ntiles) return;
    int row0 = wid * 16;
    const int Cout = NT * 16;

    v8f acc[NT];
    v8f z = {};
#pragma unroll
    for (int nt = 0; nt < NT; nt++) acc[nt] = z;

#pragma unroll
    for (int kt = 0; kt < 2; kt++) {
        v16h a = load_a_frag(act, row0, kt, lane);
#pragma unroll
        for (int nt = 0; nt < NT; nt++)
            acc[nt] = WMMA_F16(a, load_b_frag(wp, kt, nt, NT, lane), acc[nt]);
    }

    int l  = lane & 15;
    int hb = lane >> 4;
#pragma unroll
    for (int nt = 0; nt < NT; nt++)
#pragma unroll
        for (int r = 0; r < 8; r++) {
            int row = row0 + hb * 8 + r;
            outF[(size_t)row * Cout + nt * 16 + l] = acc[nt][r];
        }
}

// ---------------------------------------------------------------- GAT pieces

// per-node attention dots: ssrc[i] = h[i].a_src, sdst[i] = h[i].a_dst (wave per node)
__global__ void __launch_bounds__(256)
dot2_kernel(const float* __restrict__ h, const float* __restrict__ a_s,
            const float* __restrict__ a_d, float* __restrict__ ssrc,
            float* __restrict__ sdst, int n, int C) {
    int gid  = blockIdx.x * blockDim.x + threadIdx.x;
    int wid  = gid >> 5;
    int lane = threadIdx.x & 31;
    if (wid >= n) return;
    float vs = 0.0f, vd = 0.0f;
    for (int f = lane; f < C; f += 32) {
        float hv = h[(size_t)wid * C + f];
        vs += hv * a_s[f];
        vd += hv * a_d[f];
    }
#pragma unroll
    for (int o = 16; o > 0; o >>= 1) {
        vs += __shfl_xor(vs, o, 32);
        vd += __shfl_xor(vd, o, 32);
    }
    if (lane == 0) { ssrc[wid] = vs; sdst[wid] = vd; }
}

// fully fused GAT aggregation: per-node softmax (max, exp, denom) + weighted gather
// + bias (+relu->f16 for hidden layer, fp32 out for final). Atomic-free.
// wave per node; pass 1 = lane-parallel max; pass 2 = exp + denom + shuffle-broadcast gather.
template <int C, bool FINAL>
__global__ void __launch_bounds__(256)
gat_fused(const float* __restrict__ h, const int* __restrict__ csr_src,
          const int* __restrict__ rowstart, const float* __restrict__ ssrc,
          const float* __restrict__ sdst, const float* __restrict__ bias,
          _Float16* __restrict__ outH, float* __restrict__ outF, int n) {
    int gid  = blockIdx.x * blockDim.x + threadIdx.x;
    int wid  = gid >> 5;
    int lane = threadIdx.x & 31;
    if (wid >= n) return;
    int s0 = rowstart[wid], s1 = rowstart[wid + 1];
    float sd = sdst[wid];
    float self_logit = leaky(ssrc[wid] + sd);

    // ---- pass 1: segment max (lane-parallel over edges)
    float mx = self_logit;
    for (int j = s0 + lane; j < s1; j += 32)
        mx = fmaxf(mx, leaky(ssrc[csr_src[j]] + sd));
#pragma unroll
    for (int o = 16; o > 0; o >>= 1)
        mx = fmaxf(mx, __shfl_xor(mx, o, 32));

    // ---- pass 2: exp weights + denom + weighted feature gather
    float den = 0.0f;
    float a0 = 0.0f, a1 = 0.0f;
    for (int j0 = s0; j0 < s1; j0 += 32) {
        int   idx = 0;
        float ev  = 0.0f;
        if (j0 + lane < s1) {
            idx = csr_src[j0 + lane];
            ev  = __expf(leaky(ssrc[idx] + sd) - mx);
        }
        den += ev;
        int m = min(32, s1 - j0);
        for (int k = 0; k < m; k++) {
            int   s = __shfl(idx, k, 32);
            float w = __shfl(ev,  k, 32);
            const float* hs = h + (size_t)s * C;
            a0 += w * hs[lane];
            if (C == 64) a1 += w * hs[lane + 32];
        }
    }
#pragma unroll
    for (int o = 16; o > 0; o >>= 1)
        den += __shfl_xor(den, o, 32);

    // ---- self loop
    float wself = __expf(self_logit - mx);
    den += wself;
    const float* hi = h + (size_t)wid * C;
    a0 += wself * hi[lane];
    if (C == 64) a1 += wself * hi[lane + 32];

    float inv = 1.0f / den;
    float o0 = a0 * inv + bias[lane];
    if (FINAL) {
        outF[(size_t)wid * C + lane] = o0;
        if (C == 64) outF[(size_t)wid * C + lane + 32] = a1 * inv + bias[lane + 32];
    } else {
        o0 = o0 > 0.0f ? o0 : 0.0f;
        outH[(size_t)wid * C + lane] = (_Float16)o0;
        if (C == 64) {
            float o1 = a1 * inv + bias[lane + 32];
            o1 = o1 > 0.0f ? o1 : 0.0f;
            outH[(size_t)wid * C + lane + 32] = (_Float16)o1;
        }
    }
}

// ---------------------------------------------------------------- launch

extern "C" void kernel_launch(void* const* d_in, const int* in_sizes, int n_in,
                              void* d_out, int out_size, void* d_ws, size_t ws_size,
                              hipStream_t stream) {
    const float* x      = (const float*)d_in[0];
    const int*   ei     = (const int*)d_in[1];
    const float* W_out  = (const float*)d_in[2];
    const float* b_out  = (const float*)d_in[3];
    const float* W_root = (const float*)d_in[4];
    const float* W1     = (const float*)d_in[5];
    const float* a_s1   = (const float*)d_in[6];
    const float* a_d1   = (const float*)d_in[7];
    const float* b1     = (const float*)d_in[8];
    const float* W2     = (const float*)d_in[9];
    const float* a_s2   = (const float*)d_in[10];
    const float* a_d2   = (const float*)d_in[11];
    const float* b2     = (const float*)d_in[12];
    const int* src = ei;
    const int* dst = ei + GE;
    float* out = (float*)d_out;

    // workspace carve (256B aligned)
    char* w = (char*)d_ws;
    auto take = [&](size_t bytes) { char* p = w; w += (bytes + 255) & ~(size_t)255; return p; };
    int*      cnt      = (int*)take((size_t)GN * 4);
    int*      rowstart = (int*)take((size_t)(GN + 1) * 4);
    int*      cursor   = (int*)take((size_t)GN * 4);
    int*      csr_src  = (int*)take((size_t)GE * 4);
    float*    deg_inv  = (float*)take((size_t)GN * 4);
    _Float16* actA     = (_Float16*)take((size_t)GN * HID * 2);
    _Float16* actB     = (_Float16*)take((size_t)GN * HID * 2);
    float*    hbuf     = (float*)take((size_t)GN * HID * 4);
    float*    ssrc     = (float*)take((size_t)GN * 4);
    float*    sdst     = (float*)take((size_t)GN * 4);
    _Float16* wpack    = (_Float16*)take((size_t)14336 * 2);

    const int B = 256;
    auto cdiv = [](long a, long b) { return (int)((a + b - 1) / b); };
    const int ntiles = GN / 16;                      // 6250 (N divisible by 16)
    const int gemmG  = cdiv((long)ntiles * 32, B);
    const int nodeWG = cdiv((long)GN * 32, B);       // wave-per-node grids

    // ---- CSR build (once; shared by all 3 layers)
    zero_int<<<cdiv(GN, B), B, 0, stream>>>(cnt, GN);
    degree_count<<<cdiv(GE, B), B, 0, stream>>>(dst, cnt, GE);
    scan_kernel<<<1, 1024, 0, stream>>>(cnt, rowstart, GN);
    copy_int<<<cdiv(GN, B), B, 0, stream>>>(rowstart, cursor, GN);
    csr_fill<<<cdiv(GE, B), B, 0, stream>>>(src, dst, cursor, csr_src, GE);
    deginv_kernel<<<cdiv(GN, B), B, 0, stream>>>(cnt, deg_inv, GN);

    // ---- ClusterGCN: atomic-free gather + scale -> f16 operands
    gcn_gather<<<nodeWG, B, 0, stream>>>(x, csr_src, rowstart, deg_inv, actA, actB, GN);

    // ---- pack all weights to WMMA B-fragment layout
    pack_weights<<<cdiv(14336, B), B, 0, stream>>>(W_out, W_root, W1, W2, wpack);

    // ---- h1 = relu(aggs@W_out + x@W_root + b_out)  (fused dual-GEMM, 16 WMMAs/wave)
    gemm_gcn<<<gemmG, B, 0, stream>>>(actA, actB, wpack, wpack + 4096, b_out, actA, ntiles);

    // ---- GAT layer 1 (64 -> 64): GEMM -> dots -> fused softmax-aggregate
    gemm_gat<4><<<gemmG, B, 0, stream>>>(actA, wpack + 8192, hbuf, ntiles);
    dot2_kernel<<<nodeWG, B, 0, stream>>>(hbuf, a_s1, a_d1, ssrc, sdst, GN, HID);
    gat_fused<64, false><<<nodeWG, B, 0, stream>>>(hbuf, csr_src, rowstart, ssrc, sdst,
                                                   b1, actA, nullptr, GN);

    // ---- GAT layer 2 (64 -> 32): GEMM -> dots -> fused softmax-aggregate -> d_out
    gemm_gat<2><<<gemmG, B, 0, stream>>>(actA, wpack + 12288, hbuf, ntiles);
    dot2_kernel<<<nodeWG, B, 0, stream>>>(hbuf, a_s2, a_d2, ssrc, sdst, GN, OUTC);
    gat_fused<32, true><<<nodeWG, B, 0, stream>>>(hbuf, csr_src, rowstart, ssrc, sdst,
                                                  b2, nullptr, out, GN);
}
